// PerformerAttention_28381143891998
// MI455X (gfx1250) — matrix operand order repaired and checked by reading, not compiled, not run on hardware
//
#include <hip/hip_runtime.h>

// ---------------------------------------------------------------------------
// Performer attention for MI455X (gfx1250), bf16 WMMA pipeline with
// async global->LDS staging and double-buffered K-tiles.
// DIM=1024, H=16, D=64, F=256, B=4, T=4096.
// All B operands are stored transposed (Bt[n][k]) so every global->LDS copy
// is a contiguous 16-byte chunk (async-to-LDS friendly).
// ---------------------------------------------------------------------------

typedef __attribute__((ext_vector_type(16))) __bf16 v16bf;
typedef __attribute__((ext_vector_type(8)))  float  v8f;
typedef __attribute__((ext_vector_type(4)))  int    v4i;

union Frag16 { v16bf v; uint4 u[2]; };
union H8     { uint4 u; __bf16 h[8]; };

#if defined(__AMDGCN__) && defined(__gfx1250__) && \
    __has_builtin(__builtin_amdgcn_global_load_async_to_lds_b128) && \
    __has_builtin(__builtin_amdgcn_s_wait_asynccnt)
#define GEMM_ASYNC 1
typedef __attribute__((address_space(1))) v4i g_v4i;   // global
typedef __attribute__((address_space(3))) v4i l_v4i;   // LDS
#endif

__device__ __forceinline__ void copy16(const __bf16* __restrict__ g,
                                       __bf16* __restrict__ l)
{
#if defined(GEMM_ASYNC)
    __builtin_amdgcn_global_load_async_to_lds_b128(
        (g_v4i*)g, (l_v4i*)l, 0, 0);
#else
    *(uint4*)l = *(const uint4*)g;
#endif
}

template <int N>
__device__ __forceinline__ void wait_async()
{
#if defined(GEMM_ASYNC)
    __builtin_amdgcn_s_wait_asynccnt(N);
#endif
}

// ---------------------------------------------------------------------------
// Tiled bf16 WMMA GEMM:  C[M,N] = A[M,K] * B[K,N], B given transposed
// Bt[N][K].  f32 accumulate via v_wmma_f32_16x16x32_bf16.
//   - block tile 128(M) x 64(N); LDS K-tile 64 (2 WMMA K-steps), 2 buffers
//   - 256 threads = 8 waves, 4(M) x 2(N), each wave 32x32 (2x2 WMMA)
//   - TRANS_A:  A(m,k) = A[k*lda + m]        (kv = k'^T * v)
//   - OUT_MODE: 0 = f32, 1 = bf16, 2 = bf16 transposed (C^T[n][m], ldc = M)
//   - bMod:     B batch index = z % bMod     (proj shared across b)
// ---------------------------------------------------------------------------
template <bool TRANS_A, int OUT_MODE, bool HAS_BIAS>
__global__ __launch_bounds__(256) void gemm_wmma_bf16(
    const __bf16* __restrict__ A, const __bf16* __restrict__ Bt,
    void* __restrict__ Cv, const float* __restrict__ bias,
    int M, int N, int K, int lda, int ldbt, int ldc,
    long sA, long sB, long sC, int bMod)
{
    __shared__ __align__(16) __bf16 lA[2][128 * 64];   // [m][k]
    __shared__ __align__(16) __bf16 lB[2][64 * 64];    // [n][k]

    const int tid = threadIdx.x;
    const int m0  = blockIdx.x * 128;
    const int n0  = blockIdx.y * 64;
    const int z   = blockIdx.z;

    A  += (size_t)z * sA;
    Bt += (size_t)(bMod ? (z % bMod) : z) * sB;
    float*  Cf = (float*)Cv;
    __bf16* Cb = (__bf16*)Cv;
    const size_t cbase = (size_t)z * sC;

    const int lane = tid & 31, w = tid >> 5;
    const int wm = (w & 3) * 32;
    const int wn = (w >> 2) * 32;
    const int lg = lane >> 4;
    const int lr = lane & 15;

    v8f acc[2][2] = {};

    const int nk = K >> 6;          // K-tiles of 64

    auto prefetch = [&](int kk, int buf) {
        if (!TRANS_A) {
            #pragma unroll
            for (int it = 0; it < 4; ++it) {            // 1024 chunks of 8 bf16
                int c = tid + it * 256;
                int row = c >> 3, cc = c & 7;
                copy16(A + (size_t)(m0 + row) * lda + kk + cc * 8,
                       &lA[buf][row * 64 + cc * 8]);
            }
        } else {
            #pragma unroll
            for (int it = 0; it < 4; ++it) {            // scatter transpose
                int c  = tid + it * 256;
                int kr = c >> 4, mc = c & 15;           // 64 k-rows x 16 m-chunks
                H8 t;
                t.u = *(const uint4*)(A + (size_t)(kk + kr) * lda + m0 + mc * 8);
                #pragma unroll
                for (int j = 0; j < 8; ++j)
                    lA[buf][(mc * 8 + j) * 64 + kr] = t.h[j];
            }
        }
        #pragma unroll
        for (int it = 0; it < 2; ++it) {                // 512 chunks of 8 bf16
            int c = tid + it * 256;
            int nrow = c >> 3, cc = c & 7;
            copy16(Bt + (size_t)(n0 + nrow) * ldbt + kk + cc * 8,
                   &lB[buf][nrow * 64 + cc * 8]);
        }
    };

    constexpr int APS = TRANS_A ? 2 : 6;  // async instrs per wave per stage

    prefetch(0, 0);
    for (int kt = 0; kt < nk; ++kt) {
        const int cur = kt & 1;
        if (kt + 1 < nk) {
            prefetch((kt + 1) << 6, cur ^ 1);
            wait_async<APS>();        // drain current stage (in-order)
        } else {
            wait_async<0>();
        }
        __syncthreads();

        #pragma unroll
        for (int ks = 0; ks < 2; ++ks) {
            Frag16 a[2], b[2];
            #pragma unroll
            for (int mi = 0; mi < 2; ++mi) {
                const __bf16* p = &lA[cur][(wm + mi * 16 + lr) * 64 + ks * 32];
                a[mi].u[0] = *(const uint4*)(p + lg * 8);       // K=8*lg..+7
                a[mi].u[1] = *(const uint4*)(p + 16 + lg * 8);  // K=16+8*lg..+7
            }
            #pragma unroll
            for (int ni = 0; ni < 2; ++ni) {
                const __bf16* p =
                    &lB[cur][(wn + ni * 16 + lr) * 64 + ks * 32];
                b[ni].u[0] = *(const uint4*)(p + lg * 8);       // K=8*lg..+7
                b[ni].u[1] = *(const uint4*)(p + 16 + lg * 8);  // K=16+8*lg..+7
            }
            #pragma unroll
            for (int mi = 0; mi < 2; ++mi)
                #pragma unroll
                for (int ni = 0; ni < 2; ++ni)
                    acc[mi][ni] = __builtin_amdgcn_wmma_f32_16x16x32_bf16(
                        false, a[mi].v, false, b[ni].v,
                        (short)0, acc[mi][ni], false, false);
        }
        __syncthreads();
    }

    // epilogue: C/D layout (VGPR r -> M = r + 8*lg, N = lane%16)
    #pragma unroll
    for (int mi = 0; mi < 2; ++mi)
        #pragma unroll
        for (int ni = 0; ni < 2; ++ni) {
            const int ng = n0 + wn + ni * 16 + lr;
            const float bv = HAS_BIAS ? bias[ng] : 0.0f;
            #pragma unroll
            for (int r = 0; r < 8; ++r) {
                const int mg = m0 + wm + mi * 16 + r + 8 * lg;
                const float v = acc[mi][ni][r] + bv;
                if (OUT_MODE == 2)
                    Cb[cbase + (size_t)ng * ldc + mg] = (__bf16)v;
                else if (OUT_MODE == 1)
                    Cb[cbase + (size_t)mg * ldc + ng] = (__bf16)v;
                else
                    Cf[cbase + (size_t)mg * ldc + ng] = v;
            }
        }
}

// ---------------------------------------------------------------------------
// Elementwise helpers
// ---------------------------------------------------------------------------
__global__ void pack_bf16(const float* __restrict__ in,
                          __bf16* __restrict__ out, long n)
{
    long i = (long)blockIdx.x * 256 + threadIdx.x;
    if (i < n) out[i] = (__bf16)in[i];
}

// batched transpose-pack: in [b][K][N] f32 -> out [b][N][K] bf16
__global__ void pack_bf16_t(const float* __restrict__ in,
                            __bf16* __restrict__ out, int K, int N, long total)
{
    long i = (long)blockIdx.x * 256 + threadIdx.x;
    if (i >= total) return;
    int  n = (int)(i % N);
    long r = i / N;
    int  k = (int)(r % K);
    long b = r / K;
    out[(b * N + n) * (long)K + k] = (__bf16)in[i];
}

// QKV [16384,3072] f32 -> Qb/Kb bf16 [64][4096][64] (t-major),
//                         Vt bf16 [64][64][4096]   (d-major, GEMM-B form)
__global__ void split_qkv(const float* __restrict__ qkv,
                          __bf16* __restrict__ qb,
                          __bf16* __restrict__ kb,
                          __bf16* __restrict__ vt)
{
    size_t i = (size_t)blockIdx.x * 256 + threadIdx.x;
    int    n = (int)(i % 3072);
    size_t m = i / 3072;
    int sel = n >> 10, inner = n & 1023;
    int h = inner >> 6, d = inner & 63;
    int b = (int)(m >> 12), t = (int)(m & 4095);
    size_t bh = (size_t)b * 16 + h;
    __bf16 v = (__bf16)qkv[i];
    if (sel == 2)
        vt[(bh * 64 + d) * 4096 + t] = v;
    else
        (sel == 0 ? qb : kb)[(bh * 4096 + t) * 64 + d] = v;
}

// Per-row (256 wide) max -> exp(z - max), one wave32 per row.
__global__ __launch_bounds__(256) void rowmax_exp(
    const float* __restrict__ z, __bf16* __restrict__ out, int rows)
{
    int wave = blockIdx.x * 8 + (threadIdx.x >> 5);
    int lane = threadIdx.x & 31;
    if (wave >= rows) return;
    const float* rp = z + (size_t)wave * 256;
    float v[8], m = -1e30f;
    #pragma unroll
    for (int i = 0; i < 8; ++i) { v[i] = rp[lane + 32 * i]; m = fmaxf(m, v[i]); }
    #pragma unroll
    for (int off = 16; off > 0; off >>= 1) m = fmaxf(m, __shfl_xor(m, off, 32));
    __bf16* op = out + (size_t)wave * 256;
    #pragma unroll
    for (int i = 0; i < 8; ++i) op[lane + 32 * i] = (__bf16)__expf(v[i] - m);
}

// k_sum[bh,f] = sum_t k'[bh,t,f]
__global__ void colsum(const __bf16* __restrict__ kp, float* __restrict__ ksum)
{
    int bh = blockIdx.x, f = threadIdx.x;
    const __bf16* p = kp + (size_t)bh * 4096 * 256 + f;
    float acc = 0.0f;
    for (int t = 0; t < 4096; ++t) acc += (float)p[(size_t)t * 256];
    ksum[bh * 256 + f] = acc;
}

// dr[bh,t] = 1 / (q'[bh,t,:] . ksum[bh,:] + 1e-6)
__global__ void denom_recip(const __bf16* __restrict__ qp,
                            const float* __restrict__ ksum,
                            float* __restrict__ dr)
{
    size_t i = (size_t)blockIdx.x * 256 + threadIdx.x;   // 64*4096 rows
    int bh = (int)(i >> 12);
    const __bf16* p  = qp + i * 256;
    const float*  ks = ksum + bh * 256;
    float acc = 0.0f;
    for (int f = 0; f < 256; ++f) acc += (float)p[f] * ks[f];
    dr[i] = 1.0f / (acc + 1e-6f);
}

// Ob[b,t,h*64+d] = bf16( num[bh,t,d] * dr[bh,t] )
__global__ void finalize_mix(const float* __restrict__ num,
                             const float* __restrict__ dr,
                             __bf16* __restrict__ ob)
{
    size_t i = (size_t)blockIdx.x * 256 + threadIdx.x;   // 16384*1024
    int    c = (int)(i & 1023);
    size_t m = i >> 10;
    int h = c >> 6, d = c & 63;
    int b = (int)(m >> 12), t = (int)(m & 4095);
    size_t bh = (size_t)b * 16 + h;
    ob[i] = (__bf16)(num[(bh * 4096 + t) * 64 + d] * dr[bh * 4096 + t]);
}

// ---------------------------------------------------------------------------
// Host side
// ---------------------------------------------------------------------------
extern "C" void kernel_launch(void* const* d_in, const int* in_sizes, int n_in,
                              void* d_out, int out_size, void* d_ws, size_t ws_size,
                              hipStream_t stream)
{
    (void)in_sizes; (void)n_in; (void)out_size; (void)ws_size;
    const float* x     = (const float*)d_in[0];   // [4,4096,1024]
    const float* w_qkv = (const float*)d_in[1];   // [1024,3072]
    const float* w_out = (const float*)d_in[2];   // [1024,1024]
    const float* b_out = (const float*)d_in[3];   // [1024]
    const float* proj  = (const float*)d_in[4];   // [16,64,256]
    float* out = (float*)d_out;                   // [4,4096,1024]

    char* ws = (char*)d_ws;
    const size_t MB = 1024ull * 1024ull;
    // Layout (with overlays; peak ~653 MB):
    __bf16* Xb    = (__bf16*)(ws +   0 * MB);     // 32 MB  (later reused as Ob)
    __bf16* WqkvT = (__bf16*)(ws +  32 * MB);     //  6 MB  [3072][1024]
    __bf16* WoutT = (__bf16*)(ws +  38 * MB);     //  2 MB  [1024][1024]
    __bf16* ProjT = (__bf16*)(ws +  40 * MB);     //  0.5MB [16][256][64]
    __bf16* Qb    = (__bf16*)(ws +  41 * MB);     // 32 MB  (later reused for Num)
    __bf16* Kb    = (__bf16*)(ws +  73 * MB);     // 32 MB
    __bf16* Vt    = (__bf16*)(ws + 105 * MB);     // 32 MB  [64][64][4096]
    __bf16* Qp    = (__bf16*)(ws + 137 * MB);     // 128 MB [64][4096][256]
    __bf16* Kp    = (__bf16*)(ws + 265 * MB);     // 128 MB
    __bf16* KVt   = (__bf16*)(ws + 393 * MB);     //  2 MB  [64][64][256] (kv^T)
    float*  Ksum  = (float*) (ws + 395 * MB);     //  64 KB
    float*  DenR  = (float*) (ws + 396 * MB);     //  1 MB
    float*  Big   = (float*) (ws + 397 * MB);     // 256 MB (QKV f32 then Z f32)
    float*  Num   = (float*) (ws +  41 * MB);     // 64 MB  (overlays Qb+Kb, dead)
    __bf16* Ob    = (__bf16*)(ws +   0 * MB);     // 32 MB  (overlays Xb, dead)

    const dim3 blk(256);

    // 1) pack fp32 -> bf16 operands (A plain, B transposed)
    pack_bf16  <<<(16777216 + 255) / 256, blk, 0, stream>>>(x, Xb, 16777216L);
    pack_bf16_t<<<(3145728  + 255) / 256, blk, 0, stream>>>(w_qkv, WqkvT, 1024, 3072, 3145728L);
    pack_bf16_t<<<(1048576  + 255) / 256, blk, 0, stream>>>(w_out, WoutT, 1024, 1024, 1048576L);
    pack_bf16_t<<<(262144   + 255) / 256, blk, 0, stream>>>(proj, ProjT, 64, 256, 262144L);

    // 2) QKV = X @ Wqkv            [16384,1024]x[1024,3072] -> f32
    gemm_wmma_bf16<false, 0, false><<<dim3(128, 48, 1), blk, 0, stream>>>(
        Xb, WqkvT, Big, nullptr, 16384, 3072, 1024, 1024, 1024, 3072, 0, 0, 0, 0);
    split_qkv<<<196608, blk, 0, stream>>>(Big, Qb, Kb, Vt);

    // 3) Zq = Q @ proj[h]  (batched 64, B shared mod 16), then exp(z - max)
    gemm_wmma_bf16<false, 0, false><<<dim3(32, 4, 64), blk, 0, stream>>>(
        Qb, ProjT, Big, nullptr, 4096, 256, 64, 64, 64, 256,
        4096L * 64, 256L * 64, 4096L * 256, 16);
    rowmax_exp<<<(64 * 4096) / 8, blk, 0, stream>>>(Big, Qp, 64 * 4096);

    gemm_wmma_bf16<false, 0, false><<<dim3(32, 4, 64), blk, 0, stream>>>(
        Kb, ProjT, Big, nullptr, 4096, 256, 64, 64, 64, 256,
        4096L * 64, 256L * 64, 4096L * 256, 16);
    rowmax_exp<<<(64 * 4096) / 8, blk, 0, stream>>>(Big, Kp, 64 * 4096);

    // 4) k_sum; kv^T = (k'^T @ v)^T  (M=256,N=64,K=4096, transA, bf16^T out)
    colsum<<<64, blk, 0, stream>>>(Kp, Ksum);
    gemm_wmma_bf16<true, 2, false><<<dim3(2, 1, 64), blk, 0, stream>>>(
        Kp, Vt, KVt, nullptr, 256, 64, 4096, 256, 4096, 256,
        4096L * 256, 64L * 4096, 64L * 256, 0);

    // 5) z = 1/(q'.k_sum + eps);  num = q' @ kv  (4096x64, K=256)
    denom_recip<<<(64 * 4096) / 256, blk, 0, stream>>>(Qp, Ksum, DenR);
    gemm_wmma_bf16<false, 0, false><<<dim3(32, 1, 64), blk, 0, stream>>>(
        Qp, KVt, Num, nullptr, 4096, 64, 256, 256, 256, 64,
        4096L * 256, 64L * 256, 4096L * 64, 0);
    finalize_mix<<<16777216 / 256, blk, 0, stream>>>(Num, DenR, Ob);

    // 6) out = Ob @ Wout + b_out   [16384,1024]x[1024,1024] -> f32 d_out
    gemm_wmma_bf16<false, 0, true><<<dim3(128, 16, 1), blk, 0, stream>>>(
        Ob, WoutT, out, b_out, 16384, 1024, 1024, 1024, 1024, 1024, 0, 0, 0, 0);
}
